// MultiHeadSelfAttention_34376918238081
// MI455X (gfx1250) — matrix operand order repaired and checked by reading, not compiled, not run on hardware
//
#include <hip/hip_runtime.h>
#include <hip/hip_bf16.h>

#define D_MODEL   1024
#define NUM_HEADS 16
#define D_K       64
#define SEQ       2048
#define BATCH     4
#define M_TOT     (BATCH * SEQ)   // 8192 rows for all projections
#define PSTRIDE   40              // LDS row stride (bf16 elems) to dodge bank conflicts

typedef __attribute__((ext_vector_type(16))) __bf16 v16bf;
typedef __attribute__((ext_vector_type(8)))  __bf16 v8bf;
typedef __attribute__((ext_vector_type(8)))  float  v8f;
typedef __attribute__((ext_vector_type(4)))  unsigned u32x4;
typedef __attribute__((ext_vector_type(8)))  unsigned u32x8;

static __device__ inline v16bf make_v16(v8bf lo, v8bf hi) {
  return __builtin_shufflevector(lo, hi, 0,1,2,3,4,5,6,7,8,9,10,11,12,13,14,15);
}

// A fragment: 16x32 bf16 tile at p (row-major, leading dim lda).
// ISA layout: lanes 0-15 = rows, K 0-7 & 16-23; lanes 16-31 same rows, K 8-15 & 24-31.
static __device__ inline v16bf load_a_frag(const __bf16* p, int lda, int lane) {
  int r = lane & 15, sel = lane >> 4;
  const __bf16* q = p + r * lda + sel * 8;
  v8bf lo = *(const v8bf*)(q);
  v8bf hi = *(const v8bf*)(q + 16);
  return make_v16(lo, hi);
}

// B fragment: 32x16 bf16 tile, supplied as B^T (16 rows of 32 contiguous K vals, ld=ldb).
// ISA layout: lanes 0-15 = cols, K 0-7 & 16-23; lanes 16-31 = same cols, K 8-15 & 24-31.
static __device__ inline v16bf load_b_frag(const __bf16* bt, int ldb, int lane) {
  int nn = lane & 15, koff = (lane >> 4) * 8;
  const __bf16* q = bt + nn * ldb + koff;
  v8bf lo = *(const v8bf*)(q);
  v8bf hi = *(const v8bf*)(q + 16);
  return make_v16(lo, hi);
}

static __device__ inline v8f wmma_bf16(v16bf a, v16bf b, v8f c) {
  return __builtin_amdgcn_wmma_f32_16x16x32_bf16(false, a, false, b, (short)0, c,
                                                 false, false);
}

// Async copy of one 16-byte chunk global -> LDS (ASYNCcnt-tracked).
static __device__ inline void async_cp16(const __bf16* gptr, __bf16* lptr) {
  unsigned l = (unsigned)(uintptr_t)lptr;
  unsigned long long g = (unsigned long long)(uintptr_t)gptr;
  asm volatile("global_load_async_to_lds_b128 %0, %1, off"
               :: "v"(l), "v"(g) : "memory");
}

static __device__ inline void wait_async0() {
  asm volatile("s_wait_asynccnt 0" ::: "memory");
}

// Tensor Data Mover: 2-D tile load Global -> LDS (TENSORcnt-tracked).
// Loads tile_rows rows of tile_cols bf16 elements (row stride = row_stride_elems
// elems in memory), compacted contiguously into LDS at lds_addr.
// D# packing per CDNA5 ISA 8.3/8.4 (2D: groups 2/3 = NULL).
static __device__ inline void tdm_load_2d(const void* gaddr, unsigned lds_addr,
                                          unsigned tile_rows, unsigned tile_cols,
                                          unsigned row_stride_elems) {
  unsigned long long ga = (unsigned long long)(uintptr_t)gaddr;
  u32x4 g0;
  g0[0] = 1u;                                                  // count=1, user D#
  g0[1] = lds_addr;                                            // lds_addr [63:32]
  g0[2] = (unsigned)ga;                                        // global_addr lo
  g0[3] = (unsigned)((ga >> 32) & 0x01FFFFFFu) | (2u << 30);   // addr hi | type=2
  u32x8 g1;
  g1[0] = 1u << 16;                                            // wg_mask=0, data_size=2B
  g1[1] = (row_stride_elems & 0xFFFFu) << 16;                  // tensor_dim0[15:0]
  g1[2] = ((row_stride_elems >> 16) & 0xFFFFu)                 // tensor_dim0[31:16]
          | (0xFFFFu << 16);                                   // tensor_dim1[15:0]
  g1[3] = (tile_cols & 0xFFFFu) << 16;                         // dim1 hi=0 | tile_dim0
  g1[4] = tile_rows & 0xFFFFu;                                 // tile_dim1 | tile_dim2=0
  g1[5] = row_stride_elems;                                    // tensor_dim0_stride lo
  g1[6] = 0;                                                   // stride hi | dim1_stride lo
  g1[7] = 0;
  asm volatile("tensor_load_to_lds %0, %1" :: "s"(g0), "s"(g1) : "memory");
}

static __device__ inline void wait_tensor0() {
  asm volatile("s_wait_tensorcnt 0" ::: "memory");
}

// ---------------------------------------------------------------------------
// f32 -> bf16 conversion, 8 elems/thread
// ---------------------------------------------------------------------------
__global__ __launch_bounds__(256) void cvt_bf16_kernel(const float* __restrict__ in,
                                                       __bf16* __restrict__ out, int n8) {
  int i = blockIdx.x * blockDim.x + threadIdx.x;
  if (i >= n8) return;
  float4 f0 = ((const float4*)in)[i * 2 + 0];
  float4 f1 = ((const float4*)in)[i * 2 + 1];
  v8bf o;
  o[0] = (__bf16)f0.x; o[1] = (__bf16)f0.y; o[2] = (__bf16)f0.z; o[3] = (__bf16)f0.w;
  o[4] = (__bf16)f1.x; o[5] = (__bf16)f1.y; o[6] = (__bf16)f1.z; o[7] = (__bf16)f1.w;
  ((v8bf*)out)[i] = o;
}

// ---------------------------------------------------------------------------
// C = A[M,K] @ Bt[N,K]^T  (i.e. y = x @ W^T, W given row-major [out,in])
// Double-buffered LDS staging: A tile via global_load_async_to_lds_b128 (per
// lane), B tile via one Tensor-Data-Mover descriptor per k-step (wave 0).
// mode 0: store bf16 row-major, scaled (Q / K paths)
// mode 2: store bf16 into V^T layout [B, H, D_K, SEQ]
// mode 3: store f32 row-major (final output projection)
// Block = 256 thr = 8 waves; each wave computes a 32x32 tile; block tile 64x128.
// ---------------------------------------------------------------------------
__global__ __launch_bounds__(256) void gemm_bt_kernel(const __bf16* __restrict__ A,
                                                      const __bf16* __restrict__ Bt,
                                                      void* __restrict__ Cout,
                                                      int M, int N, int K,
                                                      int mode, float scale) {
  __shared__ __attribute__((aligned(32))) __bf16 lA[2][64 * 32];    // 2 x 4KB
  __shared__ __attribute__((aligned(32))) __bf16 lB[2][128 * 32];   // 2 x 8KB

  int tid = threadIdx.x;
  int lane = tid & 31, w = tid >> 5;
  int rb0 = blockIdx.x * 64;
  int cb0 = blockIdx.y * 128;
  int wr = (w & 1) * 32;    // wave row offset within block tile
  int wc = (w >> 1) * 32;   // wave col offset within block tile

  // fill of one k-step (32 wide) into buffer `buf`
  auto fill = [&](int buf, int kb) {
    {
      int row = tid >> 2, c = (tid & 3) * 8;                 // 64 rows x 4 chunks
      async_cp16(A + (size_t)(rb0 + row) * K + kb + c, &lA[buf][row * 32 + c]);
    }
    if (w == 0)                                              // one TDM per block
      tdm_load_2d(Bt + (size_t)cb0 * K + kb,
                  (unsigned)(uintptr_t)&lB[buf][0], 128, 32, (unsigned)K);
  };
  auto rotate = [&]() {
    wait_async0();                 // this wave's async A-chunks landed
    if (w == 0) wait_tensor0();    // TDM B tile landed (issuing wave's TENSORcnt)
    __syncthreads();               // publish to all waves / protect WAR on cur
  };

  fill(0, 0);
  rotate();

  v8f c00 = {}, c01 = {}, c10 = {}, c11 = {};
  int nk = K / 32;
  for (int ks = 0; ks < nk; ++ks) {
    int cur = ks & 1;
    if (ks + 1 < nk) fill(cur ^ 1, (ks + 1) * 32);   // overlap next fetch with compute

    v16bf a0 = load_a_frag(&lA[cur][(wr +  0) * 32], 32, lane);
    v16bf a1 = load_a_frag(&lA[cur][(wr + 16) * 32], 32, lane);
    v16bf b0 = load_b_frag(&lB[cur][(wc +  0) * 32], 32, lane);
    v16bf b1 = load_b_frag(&lB[cur][(wc + 16) * 32], 32, lane);
    c00 = wmma_bf16(a0, b0, c00);
    c01 = wmma_bf16(a0, b1, c01);
    c10 = wmma_bf16(a1, b0, c10);
    c11 = wmma_bf16(a1, b1, c11);

    rotate();
  }

  int rb = rb0 + wr, cb = cb0 + wc;
  int n = lane & 15, mb = (lane >> 4) * 8;
  v8f cc[2][2] = {{c00, c01}, {c10, c11}};
#pragma unroll
  for (int ti = 0; ti < 2; ++ti)
#pragma unroll
    for (int tj = 0; tj < 2; ++tj)
#pragma unroll
      for (int i = 0; i < 8; ++i) {
        int row = rb + ti * 16 + mb + i;
        int col = cb + tj * 16 + n;
        float v = cc[ti][tj][i] * scale;
        if (mode == 0) {
          ((__bf16*)Cout)[(size_t)row * N + col] = (__bf16)v;
        } else if (mode == 2) {
          int b = row >> 11, s = row & (SEQ - 1);
          int h = col >> 6, d = col & (D_K - 1);
          ((__bf16*)Cout)[((size_t)(b * NUM_HEADS + h) * D_K + d) * SEQ + s] = (__bf16)v;
        } else {
          ((float*)Cout)[(size_t)row * N + col] = v;
        }
      }
}

// ---------------------------------------------------------------------------
// Flash attention, causal.  Q pre-scaled by (1/sqrt(dk))*log2(e) so softmax
// runs in exp2 domain.  One wave handles 16 query rows; iterates key tiles of
// 32 columns; 16x64 f32 accumulators kept in registers; P transposed from
// C-layout to A-layout via per-wave LDS staging.
// Q,K: [B*SEQ, D_MODEL] bf16 ; Vt: [B*H, D_K, SEQ] bf16 ; O: [B*SEQ, D_MODEL] bf16
// ---------------------------------------------------------------------------
__global__ __launch_bounds__(256) void attn_kernel(const __bf16* __restrict__ Q,
                                                   const __bf16* __restrict__ Kmat,
                                                   const __bf16* __restrict__ Vt,
                                                   __bf16* __restrict__ O) {
  __shared__ __attribute__((aligned(32))) __bf16 pst[8][16 * PSTRIDE];

  int lane = threadIdx.x & 31, w = threadIdx.x >> 5;
  int bh = blockIdx.x;
  int b = bh >> 4, h = bh & 15;
  int q0 = (blockIdx.y * 8 + w) * 16;

  int n = lane & 15, mb = (lane >> 4) * 8;
  int r = lane & 15, sel = lane >> 4;

  const __bf16* qbase = Q + (size_t)(b * SEQ + q0) * D_MODEL + h * D_K;
  const __bf16* kbase = Kmat + (size_t)(b * SEQ) * D_MODEL + h * D_K;
  const __bf16* vbase = Vt + (size_t)bh * D_K * SEQ;
  __bf16* myp = &pst[w][0];

  // Q fragments for both 32-wide halves of d_k
  v16bf aq0 = load_a_frag(qbase, D_MODEL, lane);
  v16bf aq1 = load_a_frag(qbase + 32, D_MODEL, lane);

  float m2[8], l[8];
#pragma unroll
  for (int i = 0; i < 8; ++i) { m2[i] = -1e30f; l[i] = 0.0f; }
  v8f acc[4] = {};

  for (int j = 0; j < q0 + 16; j += 32) {
    // ---- scores for 32 key columns ----
    v16bf bk00 = load_b_frag(kbase + (size_t)j * D_MODEL,             D_MODEL, lane);
    v16bf bk01 = load_b_frag(kbase + (size_t)j * D_MODEL + 32,        D_MODEL, lane);
    v16bf bk10 = load_b_frag(kbase + (size_t)(j + 16) * D_MODEL,      D_MODEL, lane);
    v16bf bk11 = load_b_frag(kbase + (size_t)(j + 16) * D_MODEL + 32, D_MODEL, lane);
    v8f s0 = {}, s1 = {};
    s0 = wmma_bf16(aq0, bk00, s0);
    s0 = wmma_bf16(aq1, bk01, s0);
    s1 = wmma_bf16(aq0, bk10, s1);
    s1 = wmma_bf16(aq1, bk11, s1);

    // ---- online softmax (exp2 domain), causal mask ----
#pragma unroll
    for (int i = 0; i < 8; ++i) {
      int rowi = q0 + mb + i;
      float a0 = s0[i]; if (j + n > rowi)      a0 = -1e30f;
      float a1 = s1[i]; if (j + 16 + n > rowi) a1 = -1e30f;
      float mx = fmaxf(a0, a1);
      mx = fmaxf(mx, __shfl_xor(mx, 1));
      mx = fmaxf(mx, __shfl_xor(mx, 2));
      mx = fmaxf(mx, __shfl_xor(mx, 4));
      mx = fmaxf(mx, __shfl_xor(mx, 8));
      float mnew = fmaxf(m2[i], mx);
      float al = exp2f(m2[i] - mnew);
      float p0 = exp2f(a0 - mnew);
      float p1 = exp2f(a1 - mnew);
      float sum = p0 + p1;
      sum += __shfl_xor(sum, 1);
      sum += __shfl_xor(sum, 2);
      sum += __shfl_xor(sum, 4);
      sum += __shfl_xor(sum, 8);
      l[i] = l[i] * al + sum;
      m2[i] = mnew;
      // stage P in row-major for the A-layout reload
      myp[(mb + i) * PSTRIDE + n]      = (__bf16)p0;
      myp[(mb + i) * PSTRIDE + 16 + n] = (__bf16)p1;
      acc[0][i] *= al; acc[1][i] *= al; acc[2][i] *= al; acc[3][i] *= al;
    }

    asm volatile("s_wait_dscnt 0" ::: "memory");

    // reload P as a 16x32 A fragment
    v8bf plo = *(const v8bf*)&myp[r * PSTRIDE + sel * 8];
    v8bf phi = *(const v8bf*)&myp[r * PSTRIDE + 16 + sel * 8];
    v16bf pa = make_v16(plo, phi);

    // ---- accumulate P @ V over the 4 d_k tiles ----
#pragma unroll
    for (int t = 0; t < 4; ++t) {
      v16bf bv = load_b_frag(vbase + (size_t)(t * 16) * SEQ + j, SEQ, lane);
      acc[t] = wmma_bf16(pa, bv, acc[t]);
    }
    asm volatile("" ::: "memory");
  }

  // ---- normalize and store merged-head output (bf16) ----
  size_t orow0 = (size_t)(b * SEQ + q0);
#pragma unroll
  for (int t = 0; t < 4; ++t)
#pragma unroll
    for (int i = 0; i < 8; ++i) {
      float v = acc[t][i] / l[i];
      O[(orow0 + mb + i) * D_MODEL + h * D_K + t * 16 + n] = (__bf16)v;
    }
}

// ---------------------------------------------------------------------------
extern "C" void kernel_launch(void* const* d_in, const int* in_sizes, int n_in,
                              void* d_out, int out_size, void* d_ws, size_t ws_size,
                              hipStream_t stream) {
  const float* x  = (const float*)d_in[0];
  const float* wq = (const float*)d_in[1];
  const float* wk = (const float*)d_in[2];
  const float* wv = (const float*)d_in[3];
  const float* wo = (const float*)d_in[4];

  char* ws = (char*)d_ws;
  size_t off = 0;
  auto alloc = [&](size_t bytes) -> void* {
    void* p = ws + off;
    off += (bytes + 255) & ~(size_t)255;
    return p;
  };

  const size_t xN = (size_t)M_TOT * D_MODEL;     // 8.39M elems
  const size_t wN = (size_t)D_MODEL * D_MODEL;   // 1.05M elems

  __bf16* xb  = (__bf16*)alloc(xN * 2);
  __bf16* wqb = (__bf16*)alloc(wN * 2);
  __bf16* wkb = (__bf16*)alloc(wN * 2);
  __bf16* wvb = (__bf16*)alloc(wN * 2);
  __bf16* wob = (__bf16*)alloc(wN * 2);
  __bf16* Qb  = (__bf16*)alloc(xN * 2);
  __bf16* Kb  = (__bf16*)alloc(xN * 2);
  __bf16* Vtb = (__bf16*)alloc(xN * 2);
  __bf16* Ob  = (__bf16*)alloc(xN * 2);

  // f32 -> bf16 conversions
  {
    int n8 = (int)(xN / 8);
    cvt_bf16_kernel<<<dim3((n8 + 255) / 256), 256, 0, stream>>>(x, xb, n8);
    int w8 = (int)(wN / 8);
    dim3 gw((w8 + 255) / 256);
    cvt_bf16_kernel<<<gw, 256, 0, stream>>>(wq, wqb, w8);
    cvt_bf16_kernel<<<gw, 256, 0, stream>>>(wk, wkb, w8);
    cvt_bf16_kernel<<<gw, 256, 0, stream>>>(wv, wvb, w8);
    cvt_bf16_kernel<<<gw, 256, 0, stream>>>(wo, wob, w8);
  }

  dim3 gg(M_TOT / 64, D_MODEL / 128);
  const float qscale = 0.125f * 1.44269504f;  // 1/sqrt(64) * log2(e)
  gemm_bt_kernel<<<gg, 256, 0, stream>>>(xb, wqb, Qb,  M_TOT, D_MODEL, D_MODEL, 0, qscale);
  gemm_bt_kernel<<<gg, 256, 0, stream>>>(xb, wkb, Kb,  M_TOT, D_MODEL, D_MODEL, 0, 1.0f);
  gemm_bt_kernel<<<gg, 256, 0, stream>>>(xb, wvb, Vtb, M_TOT, D_MODEL, D_MODEL, 2, 1.0f);

  attn_kernel<<<dim3(BATCH * NUM_HEADS, SEQ / 128), 256, 0, stream>>>(Qb, Kb, Vtb, Ob);

  gemm_bt_kernel<<<gg, 256, 0, stream>>>(Ob, wob, d_out, M_TOT, D_MODEL, D_MODEL, 3, 1.0f);
}
